// TGNMemory_49134425866258
// MI455X (gfx1250) — matrix operand order repaired and checked by reading, not compiled, not run on hardware
//
#include <hip/hip_runtime.h>
#include <hip/hip_bf16.h>
#include <math.h>

typedef __attribute__((ext_vector_type(16))) _Float16 v16h;
typedef __attribute__((ext_vector_type(8)))  float    v8f;

#define Dm 128          // memory/message dim
#define K2 256          // 2*Dm (message GEMM K)
#define G3 384          // 3*Dm (GRU gate dim)

// ---------------------------------------------------------------------------
// WMMA fragment loaders for V_WMMA_F32_16X16X32_F16 (wave32).
//
// A (16x32, f16, row-major source with stride lda):
//   lane L holds row m = L & 15;  g = L >> 4
//   half e (0..15):  K = k0 + 16*g + e
//
// B (32x16):  B[k][n] = W[n][k] for gemm  C = A * W^T.
//   lane L holds column n = L & 15; g = L >> 4
//   half e (0..15):  K = k0 + 16*g + e   (16 contiguous halves -> b128 pair)
//
// C/D (16x16 f32, 8 VGPRs): lane L -> N = L&15 ; acc[v] -> M = v + 8*(L>>4)
// ---------------------------------------------------------------------------
__device__ __forceinline__ v16h load_a_frag_lds(const _Float16* A, int lda,
                                                int m, int k0, int g) {
  v16h a;
  const _Float16* row = A + m * lda + k0 + 16 * g;
#pragma unroll
  for (int e = 0; e < 16; ++e) {
    int k = e;
    a[e] = row[k];
  }
  return a;
}

__device__ __forceinline__ v16h load_b_frag_h(const _Float16* W, int ldw,
                                              int ncol, int k0, int g) {
  v16h b;
  const _Float16* row = W + (size_t)ncol * ldw + k0 + 16 * g;
#pragma unroll
  for (int e = 0; e < 16; ++e) b[e] = row[e];
  return b;
}

__device__ __forceinline__ v8f wmma_f16(v16h a, v16h b, v8f c) {
  return __builtin_amdgcn_wmma_f32_16x16x32_f16(
      false, a, false, b, (short)0, c, false, false);
}

__device__ __forceinline__ float sigmoidf_(float x) {
  return 1.0f / (1.0f + __expf(-x));
}

// ---------------------------------------------------------------------------
// Phase 0: one-shot fp32 -> f16 weight conversion.
// ---------------------------------------------------------------------------
__global__ void cvt_f16_kernel(const float* __restrict__ src,
                               _Float16* __restrict__ dst, int n) {
  int i = blockIdx.x * blockDim.x + threadIdx.x;
  if (i < n) dst[i] = (_Float16)src[i];
}

// ---------------------------------------------------------------------------
// Phase 1: per-node degree counts (float, for mean aggregation)
// ---------------------------------------------------------------------------
__global__ void count_kernel(const int* __restrict__ src,
                             const int* __restrict__ dst,
                             float* __restrict__ counts, int E) {
  int i = blockIdx.x * blockDim.x + threadIdx.x;
  if (i < E) {
    atomicAdd(&counts[src[i]], 1.0f);
    atomicAdd(&counts[dst[i]], 1.0f);
  }
}

// ---------------------------------------------------------------------------
// Phase 2: message GEMM + scatter.
// Block = 256 threads (8 waves) handles 64 edges (M = 64).
//   msg(64x128) = [src_mem | dst_mem](64x256) @ Wm^T(256x128) + bm
// Wave w owns output columns [16w, 16w+16). Per k-chunk: preload B + all 4
// A-subtile fragments (distinct regs -> ds_load clause), then 4 back-to-back
// WMMAs reusing the B fragment. 32 WMMAs/wave, 4x weight reuse.
// ---------------------------------------------------------------------------
__global__ void __launch_bounds__(256)
msg_kernel(const float* __restrict__ memory,
           const _Float16* __restrict__ WmH,
           const float* __restrict__ bm,
           const int* __restrict__ src,
           const int* __restrict__ dst,
           float* __restrict__ agg, int E) {
  __shared__ _Float16 As[64 * K2];          // 32 KB
  __shared__ int sids[64];
  __shared__ int dids[64];

  const int tid = threadIdx.x;
  const int e0  = blockIdx.x * 64;

  if (tid < 64) {
    int e = e0 + tid;
    if (e >= E) e = E - 1;                  // E is a multiple of 64 in practice
    sids[tid] = src[e];
    dids[tid] = dst[e];
  }
  __syncthreads();

  // Stage A = [src_mem | dst_mem] as f16: 64x256; 4 threads/row, 64 cols each.
  {
    const int row = tid >> 2;               // 0..63
    const int c0  = (tid & 3) << 6;         // 0,64,128,192
    const float* srow = memory + (size_t)sids[row] * Dm;
    const float* drow = memory + (size_t)dids[row] * Dm;
#pragma unroll
    for (int j = 0; j < 64; ++j) {
      int c = c0 + j;
      float v = (c < Dm) ? srow[c] : drow[c - Dm];
      As[row * K2 + c] = (_Float16)v;
    }
  }
  __syncthreads();

  const int wave = tid >> 5;                // 0..7 -> column tile
  const int lane = tid & 31;
  const int g    = lane >> 4;
  const int nn   = lane & 15;
  const int col  = wave * 16 + nn;          // output column (0..127)

  v8f acc0 = {}, acc1 = {}, acc2 = {}, acc3 = {};
#pragma unroll
  for (int kc = 0; kc < K2; kc += 32) {
    v16h b  = load_b_frag_h(WmH, K2, col, kc, g);
    v16h a0 = load_a_frag_lds(As,               K2, nn, kc, g);
    v16h a1 = load_a_frag_lds(As + 16 * K2,     K2, nn, kc, g);
    v16h a2 = load_a_frag_lds(As + 32 * K2,     K2, nn, kc, g);
    v16h a3 = load_a_frag_lds(As + 48 * K2,     K2, nn, kc, g);
    acc0 = wmma_f16(a0, b, acc0);
    acc1 = wmma_f16(a1, b, acc1);
    acc2 = wmma_f16(a2, b, acc2);
    acc3 = wmma_f16(a3, b, acc3);
  }

  const float bias = bm[col];
  v8f accs[4] = {acc0, acc1, acc2, acc3};
#pragma unroll
  for (int ms = 0; ms < 4; ++ms) {
#pragma unroll
    for (int v = 0; v < 8; ++v) {
      int   m   = ms * 16 + v + 8 * g;      // edge row within block
      float val = accs[ms][v] + bias;
      atomicAdd(&agg[(size_t)sids[m] * Dm + col], val);
      atomicAdd(&agg[(size_t)dids[m] * Dm + col], val);
    }
  }
}

// ---------------------------------------------------------------------------
// Phase 3: GRU cell for 32 nodes per block (256 threads, 8 waves).
// Wave w owns dim columns [16w,16w+16) and computes the full (r,z,n) gate
// triple for gi AND gh in registers: 6 B-column-tiles x 4 k-chunks x 2
// M-subtiles = 48 WMMAs/wave; gates fuse in registers (no G LDS, 1 barrier).
// ---------------------------------------------------------------------------
__global__ void __launch_bounds__(256)
gru_kernel(const float* __restrict__ memory,
           const _Float16* __restrict__ WihH,
           const _Float16* __restrict__ WhhH,
           const float* __restrict__ b_ih,
           const float* __restrict__ b_hh,
           const float* __restrict__ agg,
           const float* __restrict__ counts,
           float* __restrict__ out, int N) {
  __shared__ _Float16 Xs[32 * Dm];          // 8 KB mean-aggregated msgs (f16)
  __shared__ _Float16 Hs[32 * Dm];          // 8 KB current memory (f16)
  __shared__ float    Cs[32];               // per-node counts

  const int tid = threadIdx.x;              // 0..255
  const int n0  = blockIdx.x * 32;

  // Stage x and h: 32 rows x 128 cols; 8 threads/row, 16 cols each.
  {
    const int row = tid >> 3;               // 0..31
    const int c0  = (tid & 7) << 4;         // 0..112
    int node = n0 + row;
    if (node >= N) node = N - 1;            // N is a multiple of 32 in practice
    const float cnt = counts[node];
    const float inv = (cnt > 0.0f) ? (1.0f / cnt) : 0.0f;
    if ((tid & 7) == 0) Cs[row] = cnt;
    const float* arow = agg    + (size_t)node * Dm;
    const float* mrow = memory + (size_t)node * Dm;
#pragma unroll
    for (int j = 0; j < 16; ++j) {
      int c = c0 + j;
      Xs[row * Dm + c] = (_Float16)(arow[c] * inv);
      Hs[row * Dm + c] = (_Float16)mrow[c];
    }
  }
  __syncthreads();

  const int wave = tid >> 5;                // 0..7 -> dim column tile
  const int lane = tid & 31;
  const int g    = lane >> 4;
  const int nn   = lane & 15;
  const int col  = wave * 16 + nn;          // dim column (0..127)

  v8f aIr[2] = {v8f{}, v8f{}}, aIz[2] = {v8f{}, v8f{}}, aIn[2] = {v8f{}, v8f{}};
  v8f aHr[2] = {v8f{}, v8f{}}, aHz[2] = {v8f{}, v8f{}}, aHn[2] = {v8f{}, v8f{}};

#pragma unroll
  for (int kc = 0; kc < Dm; kc += 32) {
    v16h ax0 = load_a_frag_lds(Xs,            Dm, nn, kc, g);
    v16h ax1 = load_a_frag_lds(Xs + 16 * Dm,  Dm, nn, kc, g);
    v16h ah0 = load_a_frag_lds(Hs,            Dm, nn, kc, g);
    v16h ah1 = load_a_frag_lds(Hs + 16 * Dm,  Dm, nn, kc, g);

    v16h b;
    b = load_b_frag_h(WihH, Dm, col,            kc, g);   // r gate (gi)
    aIr[0] = wmma_f16(ax0, b, aIr[0]);  aIr[1] = wmma_f16(ax1, b, aIr[1]);
    b = load_b_frag_h(WihH, Dm, Dm + col,       kc, g);   // z gate (gi)
    aIz[0] = wmma_f16(ax0, b, aIz[0]);  aIz[1] = wmma_f16(ax1, b, aIz[1]);
    b = load_b_frag_h(WihH, Dm, 2 * Dm + col,   kc, g);   // n gate (gi)
    aIn[0] = wmma_f16(ax0, b, aIn[0]);  aIn[1] = wmma_f16(ax1, b, aIn[1]);
    b = load_b_frag_h(WhhH, Dm, col,            kc, g);   // r gate (gh)
    aHr[0] = wmma_f16(ah0, b, aHr[0]);  aHr[1] = wmma_f16(ah1, b, aHr[1]);
    b = load_b_frag_h(WhhH, Dm, Dm + col,       kc, g);   // z gate (gh)
    aHz[0] = wmma_f16(ah0, b, aHz[0]);  aHz[1] = wmma_f16(ah1, b, aHz[1]);
    b = load_b_frag_h(WhhH, Dm, 2 * Dm + col,   kc, g);   // n gate (gh)
    aHn[0] = wmma_f16(ah0, b, aHn[0]);  aHn[1] = wmma_f16(ah1, b, aHn[1]);
  }

  const float bir = b_ih[col],          bhr = b_hh[col];
  const float biz = b_ih[Dm + col],     bhz = b_hh[Dm + col];
  const float bin = b_ih[2 * Dm + col], bhn = b_hh[2 * Dm + col];

#pragma unroll
  for (int ms = 0; ms < 2; ++ms) {
#pragma unroll
    for (int v = 0; v < 8; ++v) {
      const int m    = ms * 16 + v + 8 * g;       // node row within block
      int node = n0 + m;
      if (node >= N) node = N - 1;
      const float cnt = Cs[m];
      const float ir = aIr[ms][v] + bir, hr = aHr[ms][v] + bhr;
      const float iz = aIz[ms][v] + biz, hz = aHz[ms][v] + bhz;
      const float in_ = aIn[ms][v] + bin, hn = aHn[ms][v] + bhn;
      const float r  = sigmoidf_(ir + hr);
      const float z  = sigmoidf_(iz + hz);
      const float nv = tanhf(in_ + r * hn);
      const float hv = memory[(size_t)node * Dm + col];   // exact f32 h (L2 hot)
      const float res = (1.0f - z) * nv + z * hv;
      out[(size_t)node * Dm + col] = (cnt > 0.0f) ? res : hv;
    }
  }
}

// ---------------------------------------------------------------------------
extern "C" void kernel_launch(void* const* d_in, const int* in_sizes, int n_in,
                              void* d_out, int out_size, void* d_ws, size_t ws_size,
                              hipStream_t stream) {
  const float* memory = (const float*)d_in[0];
  const float* Wm     = (const float*)d_in[1];
  const float* bm     = (const float*)d_in[2];
  const float* W_ih   = (const float*)d_in[3];
  const float* W_hh   = (const float*)d_in[4];
  const float* b_ih   = (const float*)d_in[5];
  const float* b_hh   = (const float*)d_in[6];
  const int*   src    = (const int*)d_in[7];
  const int*   dst    = (const int*)d_in[8];

  const int N = in_sizes[0] / Dm;   // 500,000
  const int E = in_sizes[7];        // 1,000,000

  // Workspace layout: [agg (N*Dm f32)] [counts (N f32)] [f16 weights]
  float* agg    = (float*)d_ws;
  float* counts = agg + (size_t)N * Dm;
  size_t woff   = (((size_t)N * Dm + N) * sizeof(float) + 255) & ~(size_t)255;
  _Float16* WmH  = (_Float16*)((char*)d_ws + woff);            // 128*256
  _Float16* WihH = WmH  + (size_t)Dm * K2;                     // 384*128
  _Float16* WhhH = WihH + (size_t)G3 * Dm;                     // 384*128

  hipMemsetAsync(d_ws, 0, ((size_t)N * Dm + N) * sizeof(float), stream);

  const int nWm  = Dm * K2;   // 32768
  const int nWg  = G3 * Dm;   // 49152
  cvt_f16_kernel<<<(nWm + 255) / 256, 256, 0, stream>>>(Wm,   WmH,  nWm);
  cvt_f16_kernel<<<(nWg + 255) / 256, 256, 0, stream>>>(W_ih, WihH, nWg);
  cvt_f16_kernel<<<(nWg + 255) / 256, 256, 0, stream>>>(W_hh, WhhH, nWg);

  count_kernel<<<(E + 255) / 256, 256, 0, stream>>>(src, dst, counts, E);

  const int etiles = (E + 63) / 64;
  msg_kernel<<<etiles, 256, 0, stream>>>(memory, WmH, bm, src, dst, agg, E);

  const int ntiles = (N + 31) / 32;
  gru_kernel<<<ntiles, 256, 0, stream>>>(memory, WihH, WhhH, b_ih, b_hh,
                                         agg, counts, (float*)d_out, N);
}
